// LEM_24378234372539
// MI455X (gfx1250) — matrix operand order repaired and checked by reading, not compiled, not run on hardware
//
#include <hip/hip_runtime.h>

// ---------------- problem constants ----------------
#define NINP 128
#define NHID 768
#define NOUT 128
#define TSTEPS 256
#define BATCH 256
#define BM 16              // batch rows per workgroup
#define NWG (BATCH / BM)   // 16 persistent workgroups

// WMMA tiling
#define KTI (NINP / 32)    // 4  k-tiles for x projection
#define KTH (NHID / 32)    // 24 k-tiles for hidden GEMMs
#define NTJ (NHID / 16)    // 48 n-tiles of the hidden dim

// swizzled-weight fragment counts (each fragment = 16x32 bf16 = 1KB)
#define FRAGS_INP ((4 * NHID / 16) * KTI)   // 192 * 4  = 768
#define FRAGS_HID ((3 * NHID / 16) * KTH)   // 144 * 24 = 3456
#define FRAGS_Z   ((NHID / 16) * KTH)       // 48 * 24  = 1152
#define FRAGS_CLS ((NOUT / 16) * KTH)       // 8 * 24   = 192

// workspace element offsets (in bf16 elements; 1 frag = 512 elements)
#define OFF_INP 0
#define OFF_HID (OFF_INP + (size_t)FRAGS_INP * 512)
#define OFF_Z   (OFF_HID + (size_t)FRAGS_HID * 512)
#define OFF_CLS (OFF_Z   + (size_t)FRAGS_Z   * 512)

// LDS strides (padded: f32 stride % 64 == 4 dwords, bf16 stride/2 % 64 == 4 dwords)
#define F32S 772
#define BFS  776
#define XS   136
#define SMEM_BYTES (4 * BM * F32S * 4 + 2 * BM * BFS * 2 + BM * XS * 2) // 251,648

typedef __attribute__((ext_vector_type(16))) __bf16 v16bf;
typedef __attribute__((ext_vector_type(8)))  float  v8f;

__device__ __forceinline__ v8f wmma_bf16(v16bf a, v16bf b, v8f c) {
  return __builtin_amdgcn_wmma_f32_16x16x32_bf16(
      /*neg_a=*/false, a, /*neg_b=*/false, b,
      /*c_mod=*/(short)0, c, /*reuse_a=*/false, /*reuse_b=*/false);
}

// B-operand fragment: pre-swizzled in workspace, lane-contiguous 32 bytes.
__device__ __forceinline__ v16bf load_b_frag(const __bf16* p) {
  union { v16bf v; uint4 q[2]; } f;
  f.q[0] = *(const uint4*)(p);
  f.q[1] = *(const uint4*)(p + 8);
  return f.v;
}

// A-operand fragment from row-major (padded) LDS bf16 row.
// Lane l: row = l&15, halves at K = k0+8*(l>>4) and K = k0+16+8*(l>>4).
__device__ __forceinline__ v16bf load_a_frag(const __bf16* row, int k0, int ko) {
  union { v16bf v; uint4 q[2]; } f;
  f.q[0] = *(const uint4*)(row + k0 + ko);
  f.q[1] = *(const uint4*)(row + k0 + 16 + ko);
  return f.v;
}

__device__ __forceinline__ const __bf16* bfrag(const __bf16* base, int KT, int nt,
                                               int kt, int lane) {
  return base + (((size_t)nt * KT + kt) << 9) + (lane << 4);
}

__device__ __forceinline__ float sig_f(float x) { return 1.0f / (1.0f + __expf(-x)); }
__device__ __forceinline__ float tanh_f(float x) { return 2.0f * sig_f(2.0f * x) - 1.0f; }

// ---------------- weight swizzle prep ----------------
// src: original fp32 weight [N][K] row-major. dst: fragments of B = W^T tiles.
// Fragment (nt,kt): lane l element e = src[nt*16 + (l&15)][kt*32 + (l>>4)*8 + (e&7) + (e>>3)*16]
__global__ void lem_prep_swz(const float* __restrict__ src, __bf16* __restrict__ dst,
                             int N, int K) {
  int gid = blockIdx.x * blockDim.x + threadIdx.x;
  int KT = K >> 5;
  int nfrag = (N >> 4) * KT;
  if (gid >= nfrag * 32) return;
  int frag = gid >> 5, lane = gid & 31;
  int nt = frag / KT, kt = frag - nt * KT;
  int n = nt * 16 + (lane & 15);
  int kb = kt * 32 + ((lane >> 4) << 3);
  const float* s = src + (size_t)n * K + kb;
  __bf16* d = dst + ((size_t)frag << 9) + (lane << 4);
#pragma unroll
  for (int e = 0; e < 16; ++e) d[e] = (__bf16)s[(e & 7) + ((e >> 3) << 4)];
}

// ---------------- fused persistent LEM scan ----------------
__global__ __launch_bounds__(256, 1)
void lem_persistent(const float* __restrict__ x,
                    const float* __restrict__ b_inp,
                    const float* __restrict__ b_hid,
                    const float* __restrict__ b_z,
                    const float* __restrict__ b_cls,
                    const __bf16* __restrict__ Winp,
                    const __bf16* __restrict__ Whid,
                    const __bf16* __restrict__ Wz,
                    const __bf16* __restrict__ Wcls,
                    float* __restrict__ out) {
  extern __shared__ char smem[];
  float* y32 = (float*)smem;               // [BM][F32S]
  float* z32 = y32 + BM * F32S;
  float* g1  = z32 + BM * F32S;            // ms_dt_bar
  float* izs = g1  + BM * F32S;            // i_z
  __bf16* ybf = (__bf16*)(izs + BM * F32S); // [BM][BFS]
  __bf16* zbf = ybf + BM * BFS;
  __bf16* xbf = zbf + BM * BFS;             // [BM][XS]

  const int tid  = threadIdx.x;
  const int lane = tid & 31;
  const int wave = tid >> 5;
  const int r    = lane & 15;
  const int hi   = lane >> 4;
  const int ko   = hi * 8;
  const int b0   = blockIdx.x * BM;

  for (int i = tid; i < BM * F32S; i += 256) { y32[i] = 0.0f; z32[i] = 0.0f; }
  for (int i = tid; i < BM * BFS; i += 256) { ybf[i] = (__bf16)0.0f; zbf[i] = (__bf16)0.0f; }
  __syncthreads();

  const int j0 = wave * (NTJ / 8);   // 6 n-tiles per wave
  const int j1 = j0 + (NTJ / 8);

  for (int t = 0; t < TSTEPS; ++t) {
    // ---- phase 0: stage x[t] tile (fp32 -> bf16, padded LDS) ----
    {
      const float* xp = x + ((size_t)t * BATCH + b0) * NINP;
      int idx = tid * 8;
      int row = idx >> 7, col = idx & 127;
      const float4* s = (const float4*)(xp + (size_t)row * NINP + col);
      float4 a = s[0], b = s[1];
      __bf16* d = xbf + row * XS + col;
      d[0] = (__bf16)a.x; d[1] = (__bf16)a.y; d[2] = (__bf16)a.z; d[3] = (__bf16)a.w;
      d[4] = (__bf16)b.x; d[5] = (__bf16)b.y; d[6] = (__bf16)b.z; d[7] = (__bf16)b.w;
    }
    __syncthreads();

    // ---- phase 1: xi + hid projections, z update ----
    for (int j = j0; j < j1; ++j) {
      v8f a1 = {}, a2 = {}, ay = {}, az = {};
      // input projection (K = 128): 4 output streams share one A fragment
#pragma unroll
      for (int kt = 0; kt < KTI; ++kt) {
        v16bf af = load_a_frag(xbf + r * XS, kt * 32, ko);
        v16bf w1 = load_b_frag(bfrag(Winp, KTI, j,             kt, lane));
        v16bf w2 = load_b_frag(bfrag(Winp, KTI, NTJ + j,       kt, lane));
        v16bf w3 = load_b_frag(bfrag(Winp, KTI, 2 * NTJ + j,   kt, lane));
        v16bf w4 = load_b_frag(bfrag(Winp, KTI, 3 * NTJ + j,   kt, lane));
        a1 = wmma_bf16(af, w1, a1);
        a2 = wmma_bf16(af, w2, a2);
        az = wmma_bf16(af, w3, az);
        ay = wmma_bf16(af, w4, ay);
      }
      // hidden projection (K = 768): 3 output streams share one A fragment
      for (int kt = 0; kt < KTH; ++kt) {
        v16bf af = load_a_frag(ybf + r * BFS, kt * 32, ko);
        v16bf w1 = load_b_frag(bfrag(Whid, KTH, j,           kt, lane));
        v16bf w2 = load_b_frag(bfrag(Whid, KTH, NTJ + j,     kt, lane));
        v16bf w3 = load_b_frag(bfrag(Whid, KTH, 2 * NTJ + j, kt, lane));
        if (kt + 1 < KTH) {
          __builtin_prefetch(bfrag(Whid, KTH, j, kt + 1, lane), 0, 1);
        }
        a1 = wmma_bf16(af, w1, a1);
        a2 = wmma_bf16(af, w2, a2);
        ay = wmma_bf16(af, w3, ay);
      }
      const int col = j * 16 + r;
      const float bi1 = b_inp[col], bi2 = b_inp[NHID + col];
      const float bi3 = b_inp[2 * NHID + col], bi4 = b_inp[3 * NHID + col];
      const float bh1 = b_hid[col], bh2 = b_hid[NHID + col], bh3 = b_hid[2 * NHID + col];
#pragma unroll
      for (int v = 0; v < 8; ++v) {
        const int m = v + hi * 8;
        float md_bar = sig_f(a1[v] + bi1 + bh1);          // DT = 1.0
        float md     = sig_f(a2[v] + bi2 + bh2);
        float zt     = tanh_f(ay[v] + bi4 + bh3);
        float zn = (1.0f - md) * z32[m * F32S + col] + md * zt;
        z32[m * F32S + col] = zn;
        zbf[m * BFS + col]  = (__bf16)zn;
        g1[m * F32S + col]  = md_bar;
        izs[m * F32S + col] = az[v] + bi3;
      }
    }
    __syncthreads();

    // ---- phase 2: z @ W_z^T, y update ----
    for (int j = j0; j < j1; ++j) {
      v8f acc = {};
      for (int kt = 0; kt < KTH; ++kt) {
        v16bf af = load_a_frag(zbf + r * BFS, kt * 32, ko);
        v16bf wf = load_b_frag(bfrag(Wz, KTH, j, kt, lane));
        if (kt + 1 < KTH) {
          __builtin_prefetch(bfrag(Wz, KTH, j, kt + 1, lane), 0, 1);
        }
        acc = wmma_bf16(af, wf, acc);
      }
      const int col = j * 16 + r;
      const float bz = b_z[col];
#pragma unroll
      for (int v = 0; v < 8; ++v) {
        const int m = v + hi * 8;
        float w  = acc[v] + bz + izs[m * F32S + col];
        float yt = tanh_f(w);
        float g  = g1[m * F32S + col];
        float yn = (1.0f - g) * y32[m * F32S + col] + g * yt;
        y32[m * F32S + col] = yn;
        ybf[m * BFS + col]  = (__bf16)yn;
      }
    }
    __syncthreads();
  }

  // ---- phase 3: classifier y @ W_cls^T + b_cls (one 16x16 n-tile per wave) ----
  {
    const int j = wave;   // NOUT/16 == 8 tiles == 8 waves
    v8f acc = {};
    for (int kt = 0; kt < KTH; ++kt) {
      v16bf af = load_a_frag(ybf + r * BFS, kt * 32, ko);
      v16bf wf = load_b_frag(bfrag(Wcls, KTH, j, kt, lane));
      acc = wmma_bf16(af, wf, acc);
    }
    const int col = j * 16 + r;
    const float bc = b_cls[col];
#pragma unroll
    for (int v = 0; v < 8; ++v) {
      const int m = v + hi * 8;
      out[(size_t)(b0 + m) * NOUT + col] = acc[v] + bc;
    }
  }
}

// ---------------- launcher ----------------
extern "C" void kernel_launch(void* const* d_in, const int* in_sizes, int n_in,
                              void* d_out, int out_size, void* d_ws, size_t ws_size,
                              hipStream_t stream) {
  const float* input = (const float*)d_in[0];
  const float* W_inp = (const float*)d_in[1];
  const float* b_inp = (const float*)d_in[2];
  const float* W_hid = (const float*)d_in[3];
  const float* b_hid = (const float*)d_in[4];
  const float* W_z   = (const float*)d_in[5];
  const float* b_z   = (const float*)d_in[6];
  const float* W_cls = (const float*)d_in[7];
  const float* b_cls = (const float*)d_in[8];
  float* out = (float*)d_out;

  __bf16* ws = (__bf16*)d_ws;
  __bf16* winp = ws + OFF_INP;
  __bf16* whid = ws + OFF_HID;
  __bf16* wz   = ws + OFF_Z;
  __bf16* wcls = ws + OFF_CLS;

  // one-time-per-call weight swizzle into bf16 fragment layout (ws <= 5.5 MB)
  {
    int th;
    th = FRAGS_INP * 32; lem_prep_swz<<<(th + 255) / 256, 256, 0, stream>>>(W_inp, winp, 4 * NHID, NINP);
    th = FRAGS_HID * 32; lem_prep_swz<<<(th + 255) / 256, 256, 0, stream>>>(W_hid, whid, 3 * NHID, NHID);
    th = FRAGS_Z   * 32; lem_prep_swz<<<(th + 255) / 256, 256, 0, stream>>>(W_z,   wz,   NHID,     NHID);
    th = FRAGS_CLS * 32; lem_prep_swz<<<(th + 255) / 256, 256, 0, stream>>>(W_cls, wcls, NOUT,     NHID);
  }

  (void)hipFuncSetAttribute((const void*)lem_persistent,
                            hipFuncAttributeMaxDynamicSharedMemorySize, SMEM_BYTES);
  lem_persistent<<<NWG, 256, SMEM_BYTES, stream>>>(
      input, b_inp, b_hid, b_z, b_cls, winp, whid, wz, wcls, out);
}
